// THNN_layer_notanh_90185723281664
// MI455X (gfx1250) — matrix operand order repaired and checked by reading, not compiled, not run on hardware
//
#include <hip/hip_runtime.h>

// ---------------------------------------------------------------------------
// CDNA5 (gfx1250) THNN hypergraph layer, bf16 WMMA pipeline.
// Round 2: A-fragments hoisted into registers; x staged directly from global.
// ---------------------------------------------------------------------------

typedef __attribute__((ext_vector_type(16))) __bf16 v16bf;
typedef __attribute__((ext_vector_type(8)))  __bf16 v8bf;
typedef __attribute__((ext_vector_type(8)))  float  v8f;

namespace {
constexpr int N_NODES = 100000;
constexpr int FEAT    = 256;   // bias column folded into bias vectors
constexpr int RANK    = 64;
constexpr int HID     = 512;
constexpr int OUTD    = 256;
constexpr int N_EDGE  = 200000;

// ---- workspace layout (bytes, all chunks 256B-aligned) ----
constexpr size_t SZ_EMB_B   = (size_t)N_NODES * FEAT * 2;   // bf16 embedding
constexpr size_t SZ_EMB_NEW = (size_t)N_NODES * RANK * 4;   // f32
constexpr size_t SZ_EMB2    = (size_t)N_NODES * OUTD * 4;   // f32
constexpr size_t SZ_ACC_C1  = SZ_EMB_NEW;
constexpr size_t SZ_ACC_S   = SZ_EMB2;
constexpr size_t SZ_DEG     = (((size_t)N_NODES * 4) + 255) & ~(size_t)255;
constexpr size_t SZ_WPP     = (size_t)FEAT * RANK * 2;
constexpr size_t SZ_W2AP    = (size_t)FEAT * HID  * 2;
constexpr size_t SZ_W2BP    = (size_t)HID  * OUTD * 2;
constexpr size_t SZ_WQP     = (size_t)RANK * OUTD * 2;
constexpr size_t SZ_BIASP   = 256;    // 64 f32
constexpr size_t SZ_BIAS2A  = 2048;   // 512 f32

constexpr size_t OFF_EMB_B   = 0;
constexpr size_t OFF_EMB_NEW = OFF_EMB_B   + SZ_EMB_B;
constexpr size_t OFF_EMB2    = OFF_EMB_NEW + SZ_EMB_NEW;
constexpr size_t OFF_ACC_C1  = OFF_EMB2    + SZ_EMB2;
constexpr size_t OFF_ACC_S   = OFF_ACC_C1  + SZ_ACC_C1;
constexpr size_t OFF_DEG     = OFF_ACC_S   + SZ_ACC_S;
constexpr size_t OFF_WPP     = OFF_DEG     + SZ_DEG;
constexpr size_t OFF_W2AP    = OFF_WPP     + SZ_WPP;
constexpr size_t OFF_W2BP    = OFF_W2AP    + SZ_W2AP;
constexpr size_t OFF_WQP     = OFF_W2BP    + SZ_W2BP;
constexpr size_t OFF_BIASP   = OFF_WQP     + SZ_WQP;
constexpr size_t OFF_BIAS2A  = OFF_BIASP   + SZ_BIASP;
} // namespace

// ---------------------------------------------------------------------------
// Fragment helpers (ISA 7.12.2 layouts, wave32)
// ---------------------------------------------------------------------------

__device__ __forceinline__ v16bf combine_frag(v8bf lo, v8bf hi) {
  v16bf a;
#pragma unroll
  for (int i = 0; i < 8; ++i) { a[i] = lo[i]; a[i + 8] = hi[i]; }
  return a;
}

// A-fragment (16x32 bf16) from a row-major bf16 matrix (global or LDS).
// lane 0-15: M=lane, K = kc*32 + {0..7, 16..23}; lane 16-31: K = kc*32 + {8..15, 24..31}
// `rowbase` must already point at (row m = lane&15, k = khi = (lane>>4)*8).
__device__ __forceinline__ v16bf load_afrag_at(const __bf16* rowbase, int kc) {
  const __bf16* p0 = rowbase + kc * 32;
  v8bf lo = *reinterpret_cast<const v8bf*>(p0);        // K offsets +0..7
  v8bf hi = *reinterpret_cast<const v8bf*>(p0 + 16);   // K offsets +16..23
  return combine_frag(lo, hi);
}

// B-fragment (32x16 bf16) from pre-packed weights:
// packed[((kc*nTiles + nt)*32 + lane)*16 + i] == W[kc*32 + (lane>>4)*8 + (i<8 ? i : i+8)][nt*16 + (lane&15)]
__device__ __forceinline__ v16bf load_bfrag(const __bf16* packed, int nTiles, int kc, int nt, int lane) {
  return *reinterpret_cast<const v16bf*>(packed + (((size_t)(kc * nTiles + nt) * 32 + lane) << 4));
}

__device__ __forceinline__ v8f wmma_bf16(v16bf a, v16bf b, v8f c) {
  return __builtin_amdgcn_wmma_f32_16x16x32_bf16(false, a, false, b, (short)0, c, false, false);
}

// ---------------------------------------------------------------------------
// Prep kernels
// ---------------------------------------------------------------------------

__global__ void k_convert_emb(const float* __restrict__ src, __bf16* __restrict__ dst, int n) {
  for (int i = blockIdx.x * blockDim.x + threadIdx.x; i < n; i += gridDim.x * blockDim.x)
    dst[i] = (__bf16)src[i];
}

// Pack row-major f32 weight [kRows][nCols] into WMMA B-fragment order (bf16).
__global__ void k_pack_w(const float* __restrict__ src, __bf16* __restrict__ dst,
                         int kRows, int nCols) {
  int total  = kRows * nCols;
  int nTiles = nCols >> 4;
  for (int tid = blockIdx.x * blockDim.x + threadIdx.x; tid < total;
       tid += gridDim.x * blockDim.x) {
    int i    = tid & 15;
    int lane = (tid >> 4) & 31;
    int rest = tid >> 9;
    int nt   = rest % nTiles;
    int kc   = rest / nTiles;
    int k = kc * 32 + ((lane >> 4) << 3) + (i < 8 ? i : i + 8);
    int n = nt * 16 + (lane & 15);
    dst[tid] = (__bf16)src[(size_t)k * nCols + n];
  }
}

// Fold the bias-ones row (row FEAT of Wp / W2a) into the bias vectors.
__global__ void k_bias(const float* __restrict__ Wp,  const float* __restrict__ bp,
                       const float* __restrict__ W2a, const float* __restrict__ b2a,
                       float* __restrict__ bias_p, float* __restrict__ bias_2a) {
  int t = threadIdx.x;
  if (t < RANK) bias_p[t]  = Wp[(size_t)FEAT * RANK + t] + bp[t];
  if (t < HID)  bias_2a[t] = W2a[(size_t)FEAT * HID + t] + b2a[t];
}

__global__ void k_deg(const int* __restrict__ edges, float* __restrict__ deg) {
  int i = blockIdx.x * blockDim.x + threadIdx.x;
  if (i < N_EDGE * 3) atomicAdd(&deg[edges[i]], 1.0f);
}

// ---------------------------------------------------------------------------
// Fused per-node MLP: emb_new = x@Wp+b ; emb_new2 = relu(x@W2a+b)@W2b+b2b
// One wave per 16-node tile, bf16 WMMA, f32 accumulate.
// A-fragments are register-resident: x frags straight from global (8 frags),
// h frags from LDS once (16 frags).
// ---------------------------------------------------------------------------

__global__ __launch_bounds__(64) void k_node_mlp(
    const __bf16* __restrict__ embB,  const __bf16* __restrict__ WpP,
    const __bf16* __restrict__ W2aP,  const __bf16* __restrict__ W2bP,
    const float* __restrict__ bias_p, const float* __restrict__ bias_2a,
    const float* __restrict__ b2b,
    float* __restrict__ embNew, float* __restrict__ embNew2) {
  __shared__ __bf16 lds[2 * 16 * HID];                 // 32 KB: h tiles only
  int wave = threadIdx.x >> 5;
  int lane = threadIdx.x & 31;
  int tile = blockIdx.x * 2 + wave;
  if (tile * 16 >= N_NODES) return;                    // wave-uniform guard
  int row0 = tile * 16;
  __bf16* hs = lds + wave * 16 * HID;                  // h tile [16][512]

  int col = lane & 15;
  int mhi = (lane >> 4) << 3;                          // C/D row offset (0 or 8)
  int khi = (lane >> 4) << 3;                          // A-frag K offset (0 or 8)

  // --- x A-fragments, straight from global bf16 embedding (row-major) ---
  v16bf xa[FEAT / 32];
  {
    const __bf16* xbase = embB + (size_t)(row0 + col) * FEAT + khi;
#pragma unroll
    for (int kc = 0; kc < FEAT / 32; ++kc) xa[kc] = load_afrag_at(xbase, kc);
  }

  // GEMM1: emb_new[16x64] = x @ Wp + bias_p
#pragma unroll
  for (int nt = 0; nt < RANK / 16; ++nt) {
    float b = bias_p[nt * 16 + col];
    v8f acc;
#pragma unroll
    for (int r = 0; r < 8; ++r) acc[r] = b;
#pragma unroll
    for (int kc = 0; kc < FEAT / 32; ++kc)
      acc = wmma_bf16(xa[kc], load_bfrag(WpP, RANK / 16, kc, nt, lane), acc);
#pragma unroll
    for (int r = 0; r < 8; ++r)
      embNew[(size_t)(row0 + mhi + r) * RANK + nt * 16 + col] = acc[r];
  }

  // GEMM2a: h = relu(x@W2a + bias_2a) -> LDS (bf16, row-major)
  for (int nt = 0; nt < HID / 16; ++nt) {
    float b = bias_2a[nt * 16 + col];
    v8f acc;
#pragma unroll
    for (int r = 0; r < 8; ++r) acc[r] = b;
#pragma unroll
    for (int kc = 0; kc < FEAT / 32; ++kc)
      acc = wmma_bf16(xa[kc], load_bfrag(W2aP, HID / 16, kc, nt, lane), acc);
#pragma unroll
    for (int r = 0; r < 8; ++r)
      hs[(mhi + r) * HID + nt * 16 + col] = (__bf16)fmaxf(acc[r], 0.0f);
  }

  // --- h A-fragments, loaded once from LDS (wave-local RAW via DScnt) ---
  v16bf ha[HID / 32];
  {
    const __bf16* hbase = hs + col * HID + khi;
#pragma unroll
    for (int kc = 0; kc < HID / 32; ++kc) ha[kc] = load_afrag_at(hbase, kc);
  }

  // GEMM2b: emb_new2 = h @ W2b + b2b
  for (int nt = 0; nt < OUTD / 16; ++nt) {
    float b = b2b[nt * 16 + col];
    v8f acc;
#pragma unroll
    for (int r = 0; r < 8; ++r) acc[r] = b;
#pragma unroll
    for (int kc = 0; kc < HID / 32; ++kc)
      acc = wmma_bf16(ha[kc], load_bfrag(W2bP, OUTD / 16, kc, nt, lane), acc);
#pragma unroll
    for (int r = 0; r < 8; ++r)
      embNew2[(size_t)(row0 + mhi + r) * OUTD + nt * 16 + col] = acc[r];
  }
}

// ---------------------------------------------------------------------------
// Edge kernel: one wave per edge (K=3). Leave-one-out products on RANK=64,
// relu of summed emb_new2 on OUT=256, scatter via f32 atomics.
// ---------------------------------------------------------------------------

__global__ __launch_bounds__(256) void k_edge(
    const int* __restrict__ edges,
    const float* __restrict__ embNew, const float* __restrict__ embNew2,
    const float* __restrict__ deg,
    float* __restrict__ accC1, float* __restrict__ accS) {
  int e = blockIdx.x * 8 + (threadIdx.x >> 5);
  if (e >= N_EDGE) return;
  int lane = threadIdx.x & 31;
  int v0 = edges[e * 3 + 0];
  int v1 = edges[e * 3 + 1];
  int v2 = edges[e * 3 + 2];
  float d0 = cbrtf(deg[v0]);
  float d1 = cbrtf(deg[v1]);
  float d2 = cbrtf(deg[v2]);

  // c1_j = (1/2!) * dscale_j * prod_{i != j} (dscale_i * emb_new_i)
#pragma unroll
  for (int t = 0; t < 2; ++t) {
    int r = lane + 32 * t;
    float g0 = d0 * embNew[(size_t)v0 * RANK + r];
    float g1 = d1 * embNew[(size_t)v1 * RANK + r];
    float g2 = d2 * embNew[(size_t)v2 * RANK + r];
    atomicAdd(&accC1[(size_t)v0 * RANK + r], 0.5f * d0 * g1 * g2);
    atomicAdd(&accC1[(size_t)v1 * RANK + r], 0.5f * d1 * g0 * g2);
    atomicAdd(&accC1[(size_t)v2 * RANK + r], 0.5f * d2 * g0 * g1);
  }

  // relu_s = relu(sum_j emb_new2[v_j]) scattered to all 3 nodes.
  int c = lane * 8;
#pragma unroll
  for (int t = 0; t < 2; ++t) {
    int cc = c + t * 4;
    float4 a = *reinterpret_cast<const float4*>(&embNew2[(size_t)v0 * OUTD + cc]);
    float4 b = *reinterpret_cast<const float4*>(&embNew2[(size_t)v1 * OUTD + cc]);
    float4 g = *reinterpret_cast<const float4*>(&embNew2[(size_t)v2 * OUTD + cc]);
    float sv[4];
    sv[0] = fmaxf(a.x + b.x + g.x, 0.0f);
    sv[1] = fmaxf(a.y + b.y + g.y, 0.0f);
    sv[2] = fmaxf(a.z + b.z + g.z, 0.0f);
    sv[3] = fmaxf(a.w + b.w + g.w, 0.0f);
#pragma unroll
    for (int q = 0; q < 4; ++q) {
      atomicAdd(&accS[(size_t)v0 * OUTD + cc + q], sv[q]);
      atomicAdd(&accS[(size_t)v1 * OUTD + cc + q], sv[q]);
      atomicAdd(&accS[(size_t)v2 * OUTD + cc + q], sv[q]);
    }
  }
}

// ---------------------------------------------------------------------------
// Finalize: out = relu((acc_c1/deg) @ Wq + bq + acc_s/deg).
// A-fragments built straight from f32 acc_c1 with fused 1/deg scale; no LDS.
// ---------------------------------------------------------------------------

__global__ __launch_bounds__(128) void k_finalize(
    const float* __restrict__ accC1, const float* __restrict__ accS,
    const float* __restrict__ deg, const __bf16* __restrict__ WqP,
    const float* __restrict__ bq, float* __restrict__ out) {
  int wave = threadIdx.x >> 5;
  int lane = threadIdx.x & 31;
  int tile = blockIdx.x * 4 + wave;
  if (tile * 16 >= N_NODES) return;
  int row0 = tile * 16;
  int col = lane & 15;
  int mhi = (lane >> 4) << 3;
  int khi = (lane >> 4) << 3;

  // mean_c1 A-fragments (2 k-chunks for RANK=64), scale fused into convert.
  v16bf ca[RANK / 32];
  {
    float invd = 1.0f / deg[row0 + col];
    const float* base = accC1 + (size_t)(row0 + col) * RANK + khi;
#pragma unroll
    for (int kc = 0; kc < RANK / 32; ++kc) {
      const float* p0 = base + kc * 32;
      v16bf a;
#pragma unroll
      for (int i = 0; i < 8; ++i) {
        a[i]     = (__bf16)(p0[i] * invd);
        a[i + 8] = (__bf16)(p0[16 + i] * invd);
      }
      ca[kc] = a;
    }
  }

  for (int nt = 0; nt < OUTD / 16; ++nt) {
    float b = bq[nt * 16 + col];
    v8f acc;
#pragma unroll
    for (int r = 0; r < 8; ++r) acc[r] = b;
#pragma unroll
    for (int kc = 0; kc < RANK / 32; ++kc)
      acc = wmma_bf16(ca[kc], load_bfrag(WqP, OUTD / 16, kc, nt, lane), acc);
#pragma unroll
    for (int r = 0; r < 8; ++r) {
      int row = row0 + mhi + r;
      float ms = accS[(size_t)row * OUTD + nt * 16 + col] / deg[row];
      out[(size_t)row * OUTD + nt * 16 + col] = fmaxf(acc[r] + ms, 0.0f);
    }
  }
}

// ---------------------------------------------------------------------------
// Host launcher
// ---------------------------------------------------------------------------

extern "C" void kernel_launch(void* const* d_in, const int* in_sizes, int n_in,
                              void* d_out, int out_size, void* d_ws, size_t ws_size,
                              hipStream_t stream) {
  (void)in_sizes; (void)n_in; (void)out_size; (void)ws_size;

  const float* embedding = (const float*)d_in[0];
  const int*   edges     = (const int*)  d_in[1];
  const float* Wp        = (const float*)d_in[2];
  const float* bp        = (const float*)d_in[3];
  const float* W2a       = (const float*)d_in[4];
  const float* b2a       = (const float*)d_in[5];
  const float* W2b       = (const float*)d_in[6];
  const float* b2b       = (const float*)d_in[7];
  const float* Wq        = (const float*)d_in[8];
  const float* bq        = (const float*)d_in[9];
  float* out = (float*)d_out;

  char* ws = (char*)d_ws;
  __bf16* embB    = (__bf16*)(ws + OFF_EMB_B);
  float*  embNew  = (float*) (ws + OFF_EMB_NEW);
  float*  embNew2 = (float*) (ws + OFF_EMB2);
  float*  accC1   = (float*) (ws + OFF_ACC_C1);
  float*  accS    = (float*) (ws + OFF_ACC_S);
  float*  deg     = (float*) (ws + OFF_DEG);
  __bf16* WpP     = (__bf16*)(ws + OFF_WPP);
  __bf16* W2aP    = (__bf16*)(ws + OFF_W2AP);
  __bf16* W2bP    = (__bf16*)(ws + OFF_W2BP);
  __bf16* WqP     = (__bf16*)(ws + OFF_WQP);
  float*  bias_p  = (float*) (ws + OFF_BIASP);
  float*  bias_2a = (float*) (ws + OFF_BIAS2A);

  // zero accumulators + degree (stream-ordered, graph-capture safe)
  hipMemsetAsync(accC1, 0, SZ_ACC_C1, stream);
  hipMemsetAsync(accS,  0, SZ_ACC_S,  stream);
  hipMemsetAsync(deg,   0, SZ_DEG,    stream);

  // prep
  k_convert_emb<<<8192, 256, 0, stream>>>(embedding, embB, N_NODES * FEAT);
  k_pack_w<<<64,  256, 0, stream>>>(Wp,  WpP,  FEAT, RANK);
  k_pack_w<<<512, 256, 0, stream>>>(W2a, W2aP, FEAT, HID);
  k_pack_w<<<512, 256, 0, stream>>>(W2b, W2bP, HID,  OUTD);
  k_pack_w<<<64,  256, 0, stream>>>(Wq,  WqP,  RANK, OUTD);
  k_bias<<<1, 512, 0, stream>>>(Wp, bp, W2a, b2a, bias_p, bias_2a);
  k_deg<<<(N_EDGE * 3 + 255) / 256, 256, 0, stream>>>(edges, deg);

  // per-node MLP (6250 16-row tiles, 2 waves/block)
  k_node_mlp<<<(N_NODES / 16 + 1) / 2, 64, 0, stream>>>(
      embB, WpP, W2aP, W2bP, bias_p, bias_2a, b2b, embNew, embNew2);

  // per-edge scatter (8 edges/block)
  k_edge<<<(N_EDGE + 7) / 8, 256, 0, stream>>>(edges, embNew, embNew2, deg, accC1, accS);

  // finalize (4 waves/block)
  k_finalize<<<(N_NODES / 16 + 3) / 4, 128, 0, stream>>>(accC1, accS, deg, WqP, bq, out);
}
